// get_model_58067957842353
// MI455X (gfx1250) — matrix operand order repaired and checked by reading, not compiled, not run on hardware
//
#include <hip/hip_runtime.h>

typedef __attribute__((ext_vector_type(16))) __bf16 v16bf;
typedef __attribute__((ext_vector_type(8)))  __bf16 v8bf;
typedef __attribute__((ext_vector_type(8)))  float  v8f;

#define BN_SCALE_F 0.99999500003749969f

__device__ __forceinline__ __bf16 f2bf(float f) {
  union { float f; unsigned u; } v; v.f = f;
  unsigned r = v.u + 0x7FFFu + ((v.u >> 16) & 1u);
  unsigned short h = (unsigned short)(r >> 16);
  return __builtin_bit_cast(__bf16, h);
}
__device__ __forceinline__ float bf2f(__bf16 b) {
  unsigned short h = __builtin_bit_cast(unsigned short, b);
  union { unsigned u; float f; } v; v.u = ((unsigned)h) << 16;
  return v.f;
}

// ---------------------------------------------------------------------------
// Weight prep: W[Co,Ci] f32 -> Wb[CoPad, Kpad] bf16, zero padded.
// ---------------------------------------------------------------------------
__global__ void k_prepw(const float* __restrict__ W, __bf16* __restrict__ Wb,
                        int Ci, int Co, int Kpad, int CoPad) {
  int t = blockIdx.x * blockDim.x + threadIdx.x;
  if (t >= CoPad * Kpad) return;
  int n = t / Kpad, k = t % Kpad;
  float v = (n < Co && k < Ci) ? W[(size_t)n * Ci + k] : 0.0f;
  Wb[t] = f2bf(v);
}

// ---------------------------------------------------------------------------
// WMMA GEMM: Y[M,Co] = act( X[M,Kpad](bf16) * Wb[CoPad,Kpad](bf16)^T + bias )
// Block = 128 threads = 4 waves; each wave owns a 32(M) x 64(N) tile:
// 2 A-fragments x 4 B-fragments -> 8 WMMAs per K-step of 32.
// A layout per ISA: lane m=L%16, half=L/16; elems 0..7 -> K = kb+8*half+i,
//                   elems 8..15 -> K = kb+16+8*half+i.
// B layout per ISA: lane n=L%16, half=L/16; same K order as A.
// ---------------------------------------------------------------------------
__global__ __launch_bounds__(128) void k_gemm(
    const __bf16* __restrict__ A, int lda, int Kpad,
    const __bf16* __restrict__ Wb, const float* __restrict__ bias,
    int Co,
    __bf16* __restrict__ outB, int ldb, int coStoreB,
    float* __restrict__ outF, int ldf,
    int relu, int tstore, int mOff, int nPts, int coTotal)
{
  int lane  = threadIdx.x & 31;
  int wv    = threadIdx.x >> 5;
  int mBase = (blockIdx.x * 4 + wv) * 32;
  int nBase0 = blockIdx.y * 64;
  int half = lane >> 4;
  int lm   = lane & 15;

  const __bf16* arow0 = A + (size_t)(mBase + lm) * lda;
  const __bf16* arow1 = arow0 + (size_t)16 * lda;

  v8f acc[2][4];
  #pragma unroll
  for (int u = 0; u < 2; ++u)
    #pragma unroll
    for (int t = 0; t < 4; ++t)
      #pragma unroll
      for (int v = 0; v < 8; ++v) acc[u][t][v] = 0.0f;

  for (int kb = 0; kb < Kpad; kb += 32) {
    v8bf a0lo = *(const v8bf*)(arow0 + kb + 8 * half);
    v8bf a0hi = *(const v8bf*)(arow0 + kb + 16 + 8 * half);
    v8bf a1lo = *(const v8bf*)(arow1 + kb + 8 * half);
    v8bf a1hi = *(const v8bf*)(arow1 + kb + 16 + 8 * half);
    v16bf af0, af1;
    #pragma unroll
    for (int i = 0; i < 8; ++i) {
      af0[i] = a0lo[i]; af0[i + 8] = a0hi[i];
      af1[i] = a1lo[i]; af1[i + 8] = a1hi[i];
    }
    int kB = kb + 8 * half;
    #pragma unroll
    for (int t = 0; t < 4; ++t) {
      const __bf16* brow = Wb + (size_t)(nBase0 + t * 16 + lm) * Kpad + kB;
      v8bf blo = *(const v8bf*)(brow);
      v8bf bhi = *(const v8bf*)(brow + 16);
      v16bf bfv;
      #pragma unroll
      for (int i = 0; i < 8; ++i) { bfv[i] = blo[i]; bfv[i + 8] = bhi[i]; }
      acc[0][t] = __builtin_amdgcn_wmma_f32_16x16x32_bf16(
          false, af0, false, bfv, (short)0, acc[0][t], false, false);
      acc[1][t] = __builtin_amdgcn_wmma_f32_16x16x32_bf16(
          false, af1, false, bfv, (short)0, acc[1][t], false, false);
    }
  }

  #pragma unroll
  for (int t = 0; t < 4; ++t) {
    int ng = nBase0 + t * 16 + lm;
    float bv = (ng < Co) ? bias[ng] : 0.0f;
    #pragma unroll
    for (int u = 0; u < 2; ++u) {
      #pragma unroll
      for (int v = 0; v < 8; ++v) {
        int gm = mBase + u * 16 + v + 8 * half;  // D: VGPR v -> M=v+8*half, N=lm
        float val = acc[u][t][v] + bv;
        if (relu) { val *= BN_SCALE_F; val = val > 0.0f ? val : 0.0f; }
        float sval = (ng < Co) ? val : 0.0f;
        if (outB && ng < coStoreB)
          outB[(size_t)gm * ldb + ng] = f2bf(sval);
        if (outF && ng < Co) {
          if (!tstore) {
            outF[(size_t)gm * ldf + ng] = val;
          } else {
            int gmo = mOff + gm;
            int b = gmo / nPts, n = gmo % nPts;
            outF[((size_t)b * coTotal + ng) * (size_t)nPts + n] = val;
          }
        }
      }
    }
  }
}

// ---------------------------------------------------------------------------
// Transpose (B,6,N) -> l0_xyz (B*N,3) and l0_pts (B*N,6)
// ---------------------------------------------------------------------------
__global__ void k_prep_l0(const float* __restrict__ xyz, float* __restrict__ l0xyz,
                          float* __restrict__ l0pts, int Bn, int Nn) {
  int i = blockIdx.x * blockDim.x + threadIdx.x;
  if (i >= Bn * Nn) return;
  int b = i / Nn, n = i % Nn;
  const float* src = xyz + (size_t)b * 6 * Nn;
  float v[6];
  #pragma unroll
  for (int c = 0; c < 6; ++c) v[c] = src[(size_t)c * Nn + n];
  for (int c = 0; c < 3; ++c) l0xyz[(size_t)i * 3 + c] = v[c];
  for (int c = 0; c < 6; ++c) l0pts[(size_t)i * 6 + c] = v[c];
}

// ---------------------------------------------------------------------------
// Attention head: 512 -> 512 -> 64 -> softmax(5). One block per batch.
// ---------------------------------------------------------------------------
__global__ void k_attn_head(const float* __restrict__ g,
                            const float* __restrict__ w0, const float* __restrict__ b0,
                            const float* __restrict__ w1, const float* __restrict__ b1,
                            const float* __restrict__ w2, const float* __restrict__ b2,
                            float* __restrict__ out) {
  __shared__ float sg[512];
  __shared__ float h1[512];
  __shared__ float h2[64];
  int b = blockIdx.x, tid = threadIdx.x;
  for (int i = tid; i < 512; i += blockDim.x) sg[i] = g[(size_t)b * 512 + i];
  __syncthreads();
  for (int o = tid; o < 512; o += blockDim.x) {
    const float* wr = w0 + (size_t)o * 512;
    float acc = 0.f;
    for (int k = 0; k < 512; ++k) acc += sg[k] * wr[k];
    float v = (acc + b0[o]) * BN_SCALE_F;
    h1[o] = v > 0.f ? v : 0.f;
  }
  __syncthreads();
  for (int o = tid; o < 64; o += blockDim.x) {
    const float* wr = w1 + (size_t)o * 512;
    float acc = 0.f;
    for (int k = 0; k < 512; ++k) acc += h1[k] * wr[k];
    float v = (acc + b1[o]) * BN_SCALE_F;
    h2[o] = v > 0.f ? v : 0.f;
  }
  __syncthreads();
  if (tid == 0) {
    float z[5], m = -1e30f;
    for (int j = 0; j < 5; ++j) {
      const float* wr = w2 + (size_t)j * 64;
      float acc = 0.f;
      for (int k = 0; k < 64; ++k) acc += h2[k] * wr[k];
      z[j] = acc + b2[j];
      if (z[j] > m) m = z[j];
    }
    float s = 0.f;
    for (int j = 0; j < 5; ++j) { z[j] = expf(z[j] - m); s += z[j]; }
    for (int j = 0; j < 5; ++j) out[(size_t)b * 5 + j] = z[j] / s;
  }
}

__global__ void k_attn_combine(const float* __restrict__ heads, float* __restrict__ pair, int Bn) {
  int i = blockIdx.x * blockDim.x + threadIdx.x;
  if (i >= 2 * Bn * 5) return;
  int p = i / (Bn * 5), r = i % (Bn * 5);
  pair[i] = heads[(size_t)(2 * p) * Bn * 5 + r] + heads[(size_t)(2 * p + 1) * Bn * 5 + r];
}

// ---------------------------------------------------------------------------
// Farthest point sampling. One block (256 thr) per batch; first-index argmax.
// ---------------------------------------------------------------------------
__global__ void k_fps(const float* __restrict__ xyz, int Nn, int npoint,
                      float* __restrict__ nxyz) {
  __shared__ float dist[2048];
  __shared__ float rv[256];
  __shared__ int ri[256];
  int b = blockIdx.x, tid = threadIdx.x;
  const float* px = xyz + (size_t)b * Nn * 3;
  for (int i = tid; i < Nn; i += 256) dist[i] = 1e10f;
  __syncthreads();
  int far = 0;
  for (int it = 0; it < npoint; ++it) {
    float cx = px[far * 3 + 0], cy = px[far * 3 + 1], cz = px[far * 3 + 2];
    if (tid == 0) {
      nxyz[((size_t)b * npoint + it) * 3 + 0] = cx;
      nxyz[((size_t)b * npoint + it) * 3 + 1] = cy;
      nxyz[((size_t)b * npoint + it) * 3 + 2] = cz;
    }
    float bv = -1.f; int bi = 0;
    for (int i = tid; i < Nn; i += 256) {
      float dx = px[i * 3 + 0] - cx, dy = px[i * 3 + 1] - cy, dz = px[i * 3 + 2] - cz;
      float d = dx * dx + dy * dy + dz * dz;
      float nd = fminf(dist[i], d);
      dist[i] = nd;
      if (nd > bv) { bv = nd; bi = i; }
    }
    rv[tid] = bv; ri[tid] = bi;
    __syncthreads();
    for (int s = 128; s > 0; s >>= 1) {
      if (tid < s) {
        if (rv[tid + s] > rv[tid] || (rv[tid + s] == rv[tid] && ri[tid + s] < ri[tid])) {
          rv[tid] = rv[tid + s]; ri[tid] = ri[tid + s];
        }
      }
      __syncthreads();
    }
    far = ri[0];
    __syncthreads();
  }
}

// ---------------------------------------------------------------------------
// Ball query + grouping: one block per (b,s). Selects first 36 in-range
// indices ascending, pads with first, writes bf16 feature rows [pts | gxyz | 0].
// ---------------------------------------------------------------------------
__global__ void k_group(const float* __restrict__ xyz, const float* __restrict__ pts,
                        const float* __restrict__ nxyz, int Nn, int Cp, int Kpad,
                        float r2, __bf16* __restrict__ A) {
  __shared__ int cnts[256];
  __shared__ int sel[36];
  __shared__ int stotal;
  int bs = blockIdx.x;
  int b = bs / 36;
  int tid = threadIdx.x;
  const float* px = xyz + (size_t)b * Nn * 3;
  float sx = nxyz[(size_t)bs * 3 + 0], sy = nxyz[(size_t)bs * 3 + 1], sz = nxyz[(size_t)bs * 3 + 2];
  float s2 = sx * sx + sy * sy + sz * sz;

  int chunk = (Nn + 255) / 256;
  int i0 = tid * chunk; if (i0 > Nn) i0 = Nn;
  int i1 = i0 + chunk;  if (i1 > Nn) i1 = Nn;

  int cnt = 0;
  for (int i = i0; i < i1; ++i) {
    float x = px[i * 3], y = px[i * 3 + 1], z = px[i * 3 + 2];
    float p2 = x * x + y * y + z * z;
    float dot = sx * x + sy * y + sz * z;
    float d2 = (s2 + p2) - 2.0f * dot;
    if (d2 <= r2) cnt++;
  }
  cnts[tid] = cnt;
  __syncthreads();
  for (int off = 1; off < 256; off <<= 1) {
    int v = (tid >= off) ? cnts[tid - off] : 0;
    __syncthreads();
    cnts[tid] += v;
    __syncthreads();
  }
  int rank = cnts[tid] - cnt;
  for (int i = i0; i < i1; ++i) {
    float x = px[i * 3], y = px[i * 3 + 1], z = px[i * 3 + 2];
    float p2 = x * x + y * y + z * z;
    float dot = sx * x + sy * y + sz * z;
    float d2 = (s2 + p2) - 2.0f * dot;
    if (d2 <= r2) { if (rank < 36) sel[rank] = i; rank++; }
  }
  if (tid == 0) { int t = cnts[255]; stotal = t < 36 ? t : 36; }
  __syncthreads();
  int total = stotal;
  for (int j = total + tid; j < 36; j += 256) sel[j] = sel[0];
  __syncthreads();

  for (int t = tid; t < 36 * Kpad; t += 256) {
    int j = t / Kpad, c = t % Kpad;
    int gi = sel[j];
    float v;
    if (c < Cp) v = pts[((size_t)b * Nn + gi) * Cp + c];
    else if (c < Cp + 3) {
      int cc = c - Cp;
      float s = (cc == 0) ? sx : (cc == 1 ? sy : sz);
      v = px[gi * 3 + cc] - s;
    } else v = 0.0f;
    A[(size_t)(bs * 36 + j) * Kpad + c] = f2bf(v);
  }
}

// ---------------------------------------------------------------------------
// Max over nsample + attention scale
// ---------------------------------------------------------------------------
__global__ void k_max_scale(const __bf16* __restrict__ H, int ld, int Co, int S, int ns,
                            const float* __restrict__ attn, int scaleIdx,
                            float* __restrict__ outPts, int ldo, int chOff, int Bn) {
  int i = blockIdx.x * blockDim.x + threadIdx.x;
  int total = Bn * S * Co;
  if (i >= total) return;
  int co = i % Co;
  int bs = i / Co;
  int b = bs / S;
  const __bf16* base = H + (size_t)bs * ns * ld + co;
  float m = -1e30f;
  for (int j = 0; j < ns; ++j) {
    float v = bf2f(base[(size_t)j * ld]);
    if (v > m) m = v;
  }
  float a = attn ? attn[(size_t)b * 5 + scaleIdx] : 1.0f;
  outPts[(size_t)bs * ldo + chOff + co] = m * a;
}

// ---------------------------------------------------------------------------
// 3-NN (stable top-3) + inverse-distance weights
// ---------------------------------------------------------------------------
__global__ void k_knn3(const float* __restrict__ xyz1, const float* __restrict__ xyz2,
                       int N1, int S, int* __restrict__ idx3, float* __restrict__ w3, int Bn) {
  int i = blockIdx.x * blockDim.x + threadIdx.x;
  if (i >= Bn * N1) return;
  int b = i / N1;
  float sx = xyz1[(size_t)i * 3], sy = xyz1[(size_t)i * 3 + 1], sz = xyz1[(size_t)i * 3 + 2];
  float s2 = sx * sx + sy * sy + sz * sz;
  const float* q = xyz2 + (size_t)b * S * 3;
  float d0 = 1e30f, d1 = 1e30f, d2v = 1e30f;
  int j0 = 0, j1 = 0, j2 = 0;
  for (int j = 0; j < S; ++j) {
    float x = q[j * 3], y = q[j * 3 + 1], z = q[j * 3 + 2];
    float p2 = x * x + y * y + z * z;
    float dot = sx * x + sy * y + sz * z;
    float d = (s2 + p2) - 2.0f * dot;
    if (d < d0)      { d2v = d1; j2 = j1; d1 = d0; j1 = j0; d0 = d; j0 = j; }
    else if (d < d1) { d2v = d1; j2 = j1; d1 = d;  j1 = j; }
    else if (d < d2v){ d2v = d;  j2 = j; }
  }
  float w0 = 1.0f / (d0 + 1e-8f), w1 = 1.0f / (d1 + 1e-8f), w2 = 1.0f / (d2v + 1e-8f);
  float s = w0 + w1 + w2;
  idx3[(size_t)i * 3 + 0] = j0; idx3[(size_t)i * 3 + 1] = j1; idx3[(size_t)i * 3 + 2] = j2;
  w3[(size_t)i * 3 + 0] = w0 / s; w3[(size_t)i * 3 + 1] = w1 / s; w3[(size_t)i * 3 + 2] = w2 / s;
}

// ---------------------------------------------------------------------------
// Input builders (bf16, K-padded)
// ---------------------------------------------------------------------------
__global__ void k_build_sa3(const float* __restrict__ l2xyz, const float* __restrict__ l2pts,
                            __bf16* __restrict__ A, int rows, int Kpad) {
  int t = blockIdx.x * blockDim.x + threadIdx.x;
  if (t >= rows * Kpad) return;
  int r = t / Kpad, c = t % Kpad;
  float v;
  if (c < 3) v = l2xyz[(size_t)r * 3 + c];
  else if (c < 627) v = l2pts[(size_t)r * 624 + (c - 3)];
  else v = 0.0f;
  A[t] = f2bf(v);
}

__global__ void k_build_fp3(const float* __restrict__ l2pts, const float* __restrict__ l3pts,
                            __bf16* __restrict__ A, int rows, int Kpad) {
  int t = blockIdx.x * blockDim.x + threadIdx.x;
  if (t >= rows * Kpad) return;
  int r = t / Kpad, c = t % Kpad;
  int b = r / 36;
  float v;
  if (c < 624) v = l2pts[(size_t)r * 624 + c];
  else if (c < 1648) v = l3pts[(size_t)b * 1024 + (c - 624)];
  else v = 0.0f;
  A[t] = f2bf(v);
}

__global__ void k_build_fp2(const float* __restrict__ l1pts, const float* __restrict__ fp3o,
                            const int* __restrict__ idx3, const float* __restrict__ w3,
                            __bf16* __restrict__ A, int rows, int Kpad) {
  int t = blockIdx.x * blockDim.x + threadIdx.x;
  if (t >= rows * Kpad) return;
  int r = t / Kpad, c = t % Kpad;
  int b = r / 36;
  float v;
  if (c < 624) v = l1pts[(size_t)r * 624 + c];
  else if (c < 1136) {
    int cc = c - 624;
    v = 0.0f;
    for (int k = 0; k < 3; ++k)
      v += w3[(size_t)r * 3 + k] * fp3o[((size_t)b * 36 + idx3[(size_t)r * 3 + k]) * 512 + cc];
  } else v = 0.0f;
  A[t] = f2bf(v);
}

__global__ void k_build_fp1(const float* __restrict__ g, const float* __restrict__ l0xyz,
                            const float* __restrict__ l0pts, const float* __restrict__ fp2o,
                            const int* __restrict__ idx3, const float* __restrict__ w3,
                            __bf16* __restrict__ A, int mBase, int rows, int Kpad) {
  int t = blockIdx.x * blockDim.x + threadIdx.x;
  if (t >= rows * Kpad) return;
  int r = t / Kpad, c = t % Kpad;
  int m = mBase + r;
  int b = m / 2048;
  float v;
  if (c < 512) v = g[(size_t)b * 512 + c];
  else if (c < 515) v = l0xyz[(size_t)m * 3 + (c - 512)];
  else if (c < 521) v = l0pts[(size_t)m * 6 + (c - 515)];
  else if (c < 905) {
    int cc = c - 521;
    v = 0.0f;
    for (int k = 0; k < 3; ++k)
      v += w3[(size_t)m * 3 + k] * fp2o[((size_t)b * 36 + idx3[(size_t)m * 3 + k]) * 384 + cc];
  } else v = 0.0f;
  A[t] = f2bf(v);
}

// ===========================================================================
extern "C" void kernel_launch(void* const* d_in, const int* in_sizes, int n_in,
                              void* d_out, int out_size, void* d_ws, size_t ws_size,
                              hipStream_t stream) {
  (void)in_sizes; (void)n_in; (void)out_size; (void)ws_size;
  const int Bn = 32, Nn = 2048, NP = 36, NS = 36, C1 = 624;
  const float* xyz = (const float*)d_in[0];
  const float* g   = (const float*)d_in[1];
  auto F = [&](int i) { return (const float*)d_in[i]; };
  // Pytree order (sorted keys): attn[h]{b0,b1,b2,w0,w1,w2} h=0..3 -> 2+6h;
  // conv1{b0,w0}->26; fp1{b0,b1,w0,w1}->28; fp2->32; fp3->36;
  // sa1[s]{b0,b1,w0,w1}->40+4s; sa2->60+4s; sa3->80.

  size_t off = 0;
  auto alloc = [&](size_t bytes) -> char* {
    off = (off + 255) & ~(size_t)255;
    char* p = (char*)d_ws + off;
    off += bytes;
    return p;
  };
  float* l0xyz = (float*)alloc((size_t)Bn * Nn * 3 * 4);
  float* l0pts = (float*)alloc((size_t)Bn * Nn * 6 * 4);
  float* attnH = (float*)alloc((size_t)4 * Bn * 5 * 4);
  float* attnP = (float*)alloc((size_t)2 * Bn * 5 * 4);
  float* l1xyz = (float*)alloc((size_t)Bn * NP * 3 * 4);
  float* l2xyz = (float*)alloc((size_t)Bn * NP * 3 * 4);
  float* l1pts = (float*)alloc((size_t)Bn * NP * C1 * 4);
  float* l2pts = (float*)alloc((size_t)Bn * NP * C1 * 4);
  float* l3pts = (float*)alloc((size_t)Bn * 1024 * 4);
  float* fp3o  = (float*)alloc((size_t)Bn * NP * 512 * 4);
  float* fp2o  = (float*)alloc((size_t)Bn * NP * 384 * 4);
  int*   kidx2 = (int*)alloc((size_t)Bn * NP * 3 * 4);
  float* kw2   = (float*)alloc((size_t)Bn * NP * 3 * 4);
  int*   kidx1 = (int*)alloc((size_t)Bn * Nn * 3 * 4);
  float* kw1   = (float*)alloc((size_t)Bn * Nn * 3 * 4);
  __bf16* Abuf = (__bf16*)alloc((size_t)41472 * 640 * 2);
  __bf16* H1   = (__bf16*)alloc((size_t)41472 * 192 * 2);
  __bf16* H2   = (__bf16*)alloc((size_t)41472 * 256 * 2);
  __bf16* Wbuf = (__bf16*)alloc((size_t)1024 * 1664 * 2);  // largest: fp3 w0

  // prep weights into Wbuf, then GEMM (stream order makes Wbuf reuse safe)
  auto gemm = [&](const __bf16* A, int lda, int Kpad, const float* Wt, const float* bias,
                  int Ci, int Co, int M, __bf16* oB, int ldb, int coStB,
                  float* oF, int ldf, int relu, int tstore, int mOff) {
    int CoPad = ((Co + 63) / 64) * 64;
    int wtot = CoPad * Kpad;
    k_prepw<<<(wtot + 255) / 256, 256, 0, stream>>>(Wt, Wbuf, Ci, Co, Kpad, CoPad);
    int maxCo = Co > coStB ? Co : coStB;
    dim3 gr(M / 128, (maxCo + 63) / 64);
    k_gemm<<<gr, 128, 0, stream>>>(A, lda, Kpad, Wbuf, bias, Co, oB, ldb, coStB,
                                   oF, ldf, relu, tstore, mOff, Nn, 32);
  };

  // ---- l0 transpose ----
  k_prep_l0<<<(Bn * Nn + 255) / 256, 256, 0, stream>>>(xyz, l0xyz, l0pts, Bn, Nn);

  // ---- attention heads ----
  for (int h = 0; h < 4; ++h) {
    int base = 2 + 6 * h;
    k_attn_head<<<Bn, 256, 0, stream>>>(g, F(base + 3), F(base + 0), F(base + 4),
                                        F(base + 1), F(base + 5), F(base + 2),
                                        attnH + (size_t)h * Bn * 5);
  }
  k_attn_combine<<<(2 * Bn * 5 + 255) / 256, 256, 0, stream>>>(attnH, attnP, Bn);

  // ---- SA1 ----
  k_fps<<<Bn, 256, 0, stream>>>(l0xyz, Nn, NP, l1xyz);
  static const int c0_1[5] = {64, 128, 64, 32, 24}, c1_1[5] = {128, 256, 128, 64, 48};
  static const int c0p_1[5] = {64, 128, 64, 32, 32};
  static const float R2[5] = {0.01f, 0.04f, 0.09f, 0.36f, 1.0f};
  static const int choff[5] = {0, 128, 384, 512, 576};
  int M1 = Bn * NP * NS;  // 41472
  for (int s = 0; s < 5; ++s) {
    k_group<<<Bn * NP, 256, 0, stream>>>(l0xyz, l0pts, l1xyz, Nn, 6, 32, R2[s], Abuf);
    int wi = 40 + 4 * s;
    gemm(Abuf, 32, 32, F(wi + 2), F(wi + 0), 9, c0_1[s], M1, H1, c0p_1[s], c0p_1[s], nullptr, 0, 1, 0, 0);
    gemm(H1, c0p_1[s], c0p_1[s], F(wi + 3), F(wi + 1), c0_1[s], c1_1[s], M1, H2, c1_1[s], c1_1[s], nullptr, 0, 1, 0, 0);
    k_max_scale<<<(Bn * NP * c1_1[s] + 255) / 256, 256, 0, stream>>>(
        H2, c1_1[s], c1_1[s], NP, NS, attnP + 0 * Bn * 5, s, l1pts, C1, choff[s], Bn);
  }

  // ---- SA2 ----
  k_fps<<<Bn, 256, 0, stream>>>(l1xyz, NP, NP, l2xyz);
  static const int c0_2[5] = {96, 192, 96, 48, 36}, c1_2[5] = {128, 256, 128, 64, 48};
  static const int c0p_2[5] = {96, 192, 96, 64, 64};
  for (int s = 0; s < 5; ++s) {
    k_group<<<Bn * NP, 256, 0, stream>>>(l1xyz, l1pts, l2xyz, NP, C1, 640, R2[s], Abuf);
    int wi = 60 + 4 * s;
    gemm(Abuf, 640, 640, F(wi + 2), F(wi + 0), 627, c0_2[s], M1, H1, c0p_2[s], c0p_2[s], nullptr, 0, 1, 0, 0);
    gemm(H1, c0p_2[s], c0p_2[s], F(wi + 3), F(wi + 1), c0_2[s], c1_2[s], M1, H2, c1_2[s], c1_2[s], nullptr, 0, 1, 0, 0);
    k_max_scale<<<(Bn * NP * c1_2[s] + 255) / 256, 256, 0, stream>>>(
        H2, c1_2[s], c1_2[s], NP, NS, attnP + 1 * Bn * 5, s, l2pts, C1, choff[s], Bn);
  }

  // ---- SA3 (group all) ----
  {
    int rows = Bn * NP;
    k_build_sa3<<<(rows * 640 + 255) / 256, 256, 0, stream>>>(l2xyz, l2pts, Abuf, rows, 640);
    gemm(Abuf, 640, 640, F(82), F(80), 627, 300, rows, H1, 320, 320, nullptr, 0, 1, 0, 0);
    gemm(H1, 320, 320, F(83), F(81), 300, 1024, rows, H2, 1024, 1024, nullptr, 0, 1, 0, 0);
    k_max_scale<<<(Bn * 1024 + 255) / 256, 256, 0, stream>>>(H2, 1024, 1024, 1, NS, nullptr, 0, l3pts, 1024, 0, Bn);
  }

  // ---- FP3 ----
  {
    int rows = Bn * NP;
    k_build_fp3<<<(rows * 1664 + 255) / 256, 256, 0, stream>>>(l2pts, l3pts, Abuf, rows, 1664);
    gemm(Abuf, 1664, 1664, F(38), F(36), 1648, 1024, rows, H1, 1024, 1024, nullptr, 0, 1, 0, 0);
    gemm(H1, 1024, 1024, F(39), F(37), 1024, 512, rows, nullptr, 0, 0, fp3o, 512, 1, 0, 0);
  }

  // ---- FP2 ----
  {
    int rows = Bn * NP;
    k_knn3<<<(Bn * NP + 255) / 256, 256, 0, stream>>>(l1xyz, l2xyz, NP, NP, kidx2, kw2, Bn);
    k_build_fp2<<<(rows * 1152 + 255) / 256, 256, 0, stream>>>(l1pts, fp3o, kidx2, kw2, Abuf, rows, 1152);
    gemm(Abuf, 1152, 1152, F(34), F(32), 1136, 512, rows, H1, 512, 512, nullptr, 0, 1, 0, 0);
    gemm(H1, 512, 512, F(35), F(33), 512, 384, rows, nullptr, 0, 0, fp2o, 384, 1, 0, 0);
  }

  // ---- FP1 + conv1 (tiled over M) ----
  k_knn3<<<(Bn * Nn + 255) / 256, 256, 0, stream>>>(l0xyz, l1xyz, Nn, NP, kidx1, kw1, Bn);
  float* out = (float*)d_out;
  for (int tile = 0; tile < 8; ++tile) {
    int mBase = tile * 8192, rows = 8192;
    k_build_fp1<<<(rows * 928 + 255) / 256, 256, 0, stream>>>(g, l0xyz, l0pts, fp2o, kidx1, kw1, Abuf, mBase, rows, 928);
    gemm(Abuf, 928, 928, F(30), F(28), 905, 512, rows, H1, 512, 512, nullptr, 0, 1, 0, 0);
    gemm(H1, 512, 512, F(31), F(29), 512, 256, rows, H2, 256, 256, nullptr, 0, 1, 0, 0);
    gemm(H2, 256, 256, F(27), F(26), 256, 32, rows, nullptr, 0, 0, out, 0, 0, 1, mBase);
  }
}
